// GUNET_83854941487721
// MI455X (gfx1250) — compile-verified, matrix operand order checked by reading, not run on hardware
//
#include <hip/hip_runtime.h>

// ---------------------------------------------------------------------------
// Graph U-Net for MI455X (gfx1250, wave32).
// Dense x@W products run on V_WMMA_F32_16X16X32_BF16 (bf16 in, f32 acc):
//  - W pre-swizzled into per-lane fragment order (B frag = one 32B load,
//    hoisted into VGPRs, reused across row tiles)
//  - A tiles DMA'd into double-buffered LDS by the Tensor Data Mover
//    (tensor_load_to_lds + s_wait_tensorcnt), overlapping WMMA of tile t
//    with the DMA of tile t+1. TDM OOB-zero-fill handles the ragged tail.
// Edge aggregation is f32 atomic scatter (L2-resident: xw <= 51MB << 192MB L2).
// ---------------------------------------------------------------------------

typedef __bf16 bf16_t;
typedef __attribute__((ext_vector_type(16))) __bf16 v16bf;
typedef __attribute__((ext_vector_type(8)))  __bf16 v8bf;
typedef __attribute__((ext_vector_type(8)))  float  v8f;
typedef unsigned int u32x4 __attribute__((ext_vector_type(4)));
typedef int          i32x4 __attribute__((ext_vector_type(4)));
typedef int          i32x8 __attribute__((ext_vector_type(8)));

union v16u { v16bf v; v8bf h[2]; };

#define TPB 256

#if defined(__has_builtin)
#  if __has_builtin(__builtin_amdgcn_tensor_load_to_lds)
#    define HAVE_TDM 1
#  endif
#endif
#ifndef HAVE_TDM
#  define HAVE_TDM 0
#endif

// ---------------- TDM: 2D tile (rows x Kpad bf16) -> LDS --------------------
#if HAVE_TDM
__device__ __forceinline__
void tdm_load_2d(unsigned lds_off, const void* gaddr,
                 int tensor_d0, int tensor_d1, int tile_d0, int tile_d1,
                 int d0_stride) {
    unsigned long long ga = (unsigned long long)(uintptr_t)gaddr;
    u32x4 g0;
    g0[0] = 1u;                                        // count=1 (valid), user mode
    g0[1] = lds_off;                                   // lds_addr (bytes)
    g0[2] = (unsigned)(ga & 0xffffffffu);              // global_addr[31:0]
    g0[3] = (unsigned)((ga >> 32) & 0x1ffffffu)        // global_addr[56:32]
          | (2u << 30);                                // type = 2 ("image")
    i32x8 g1;
    g1[0] = (1 << 16);                                 // data_size=1 -> 2 bytes
    g1[1] = (tensor_d0 & 0xffff) << 16;                // tensor_dim0[15:0]
    g1[2] = ((tensor_d0 >> 16) & 0xffff)               // tensor_dim0[31:16]
          | ((tensor_d1 & 0xffff) << 16);              // tensor_dim1[15:0]
    g1[3] = ((tensor_d1 >> 16) & 0xffff)               // tensor_dim1[31:16]
          | ((tile_d0 & 0xffff) << 16);                // tile_dim0
    g1[4] = (tile_d1 & 0xffff);                        // tile_dim1 (tile_dim2=0)
    g1[5] = d0_stride;                                 // tensor_dim0_stride[31:0]
    g1[6] = 0;                                         // stride[47:32], dim1_stride lo
    g1[7] = 0;
    i32x4 z4 = {0, 0, 0, 0};
#if __has_include(<hip/amd_detail/amd_gfx1250_TDM.h>)
    i32x8 z8 = {0, 0, 0, 0, 0, 0, 0, 0};
    __builtin_amdgcn_tensor_load_to_lds(g0, g1, z4, z4, z8, 0);
#else
    __builtin_amdgcn_tensor_load_to_lds(g0, g1, z4, z4, 0);
#endif
}
#endif

// ---------------- elementwise / graph-structure kernels ---------------------

__global__ void k_fill_f32(float* __restrict__ p, float v, long n) {
    long t = (long)blockIdx.x * blockDim.x + threadIdx.x;
    if (t < n) p[t] = v;
}

__global__ void k_reindex(const int* __restrict__ src, const int* __restrict__ cluster,
                          int* __restrict__ dst, long n) {
    long t = (long)blockIdx.x * blockDim.x + threadIdx.x;
    if (t < n) dst[t] = cluster[src[t]];
}

__global__ void k_deg_count(const int* __restrict__ row, float* __restrict__ deg, int E) {
    long t = (long)blockIdx.x * blockDim.x + threadIdx.x;
    if (t < E) atomicAdd(&deg[row[t]], 1.0f);
}

__global__ void k_rsqrt_ip(float* __restrict__ p, int n) {
    long t = (long)blockIdx.x * blockDim.x + threadIdx.x;
    if (t < n) p[t] = rsqrtf(p[t]);
}

__global__ void k_pool_count(const int* __restrict__ cluster, float* __restrict__ cnt, int n) {
    long t = (long)blockIdx.x * blockDim.x + threadIdx.x;
    if (t < n) atomicAdd(&cnt[cluster[t]], 1.0f);
}

__global__ void k_relu_ip(float* __restrict__ p, long n) {
    long t = (long)blockIdx.x * blockDim.x + threadIdx.x;
    if (t < n) p[t] = fmaxf(p[t], 0.0f);
}

__global__ void k_sigmoid(const float* __restrict__ in, float* __restrict__ out, int n) {
    long t = (long)blockIdx.x * blockDim.x + threadIdx.x;
    if (t < n) out[t] = 1.0f / (1.0f + expf(-in[t]));
}

// ---------------- bf16 operand preparation ----------------------------------

// dst[n, Kpad] = bf16( [src0 (w0 cols) | src1 (w1 cols) | zero-pad] )
__global__ void k_cast2_bf16(const float* __restrict__ s0, int w0,
                             const float* __restrict__ s1, int w1,
                             bf16_t* __restrict__ dst, int n, int Kpad) {
    long t = (long)blockIdx.x * blockDim.x + threadIdx.x;
    if (t >= (long)n * Kpad) return;
    int r = (int)(t / Kpad), c = (int)(t % Kpad);
    float v = 0.0f;
    if (c < w0)            v = s0[(size_t)r * w0 + c];
    else if (c < w0 + w1)  v = s1[(size_t)r * w1 + (c - w0)];
    dst[t] = (bf16_t)v;
}

// Pre-swizzle W[Krows,128] (f32, row-major) into bf16 per-lane fragment order:
//   Wp[((ct*KSTEPS + ks)*32 + lane)*16 + i] = W[k, ct*16 + (lane&15)]
//   with k = ks*32 + ((i&8)<<1) + ((lane>>4)<<3) + (i&7), zero-padded for k>=Krows.
__global__ void k_packW_bf16(const float* __restrict__ W, int Krows,
                             bf16_t* __restrict__ Wp, int Kpad) {
    long t = (long)blockIdx.x * blockDim.x + threadIdx.x;
    if (t >= (long)Kpad * 128) return;
    int i      = (int)(t & 15);
    int lane   = (int)((t >> 4) & 31);
    int grp    = (int)(t >> 9);           // ct*KSTEPS + ks
    int ksteps = Kpad >> 5;
    int ct = grp / ksteps;
    int ks = grp - ct * ksteps;
    int k  = (ks << 5) + ((i & 8) << 1) + ((lane >> 4) << 3) + (i & 7);
    int col = (ct << 4) + (lane & 15);
    float v = (k < Krows) ? W[(size_t)k * 128 + col] : 0.0f;
    Wp[t] = (bf16_t)v;
}

// ---------------- WMMA GEMM: C[n,128] = A[n,KPAD] @ W[KPAD,128] -------------
// Block = 256 threads = 8 waves; wave w owns output cols [16w, 16w+16).
// 4 row-tiles per block; A tiles TDM-prefetched into double-buffered LDS.

template <int KPAD>
__global__ void __launch_bounds__(TPB)
k_wmma_gemm_bf16(const bf16_t* __restrict__ A, const bf16_t* __restrict__ Wp,
                 float* __restrict__ C, int n) {
    constexpr int KSTEPS = KPAD / 32;
    constexpr int TILES  = 4;
    __shared__ bf16_t sA[2][16 * KPAD];

    const int tid  = threadIdx.x;
    const int wave = tid >> 5;
    const int lane = tid & 31;
    const int blk0 = blockIdx.x * (16 * TILES);

    __builtin_prefetch(Wp + (size_t)wave * KSTEPS * 32 * 16, 0, 3);

    const int mlane = lane & 15;
    const int khalf = lane >> 4;
    const int col   = (wave << 4) + mlane;
    const bf16_t* wp = Wp + ((size_t)wave * KSTEPS * 32 + lane) * 16;

    // B fragments are invariant across row tiles: hoist into VGPRs once.
    v16u bfrag[KSTEPS];
#pragma unroll
    for (int ks = 0; ks < KSTEPS; ++ks)
        bfrag[ks].v = *(const v16bf*)(wp + (size_t)ks * 32 * 16);

#if HAVE_TDM
    if (wave == 0) {  // kick off DMA of tile 0 (grid sizing guarantees validity)
        tdm_load_2d((unsigned)(uintptr_t)&sA[0][0], A + (size_t)blk0 * KPAD,
                    KPAD, n - blk0, KPAD, 16, KPAD);
    }
#endif

#pragma unroll
    for (int t = 0; t < TILES; ++t) {
        const int row0 = blk0 + t * 16;
        if (row0 >= n) break;                       // uniform across block
        const int  rnext   = blk0 + (t + 1) * 16;
        const bool more    = (t + 1 < TILES) && (rnext < n);

#if HAVE_TDM
        if (wave == 0) {
            if (more) {
                tdm_load_2d((unsigned)(uintptr_t)&sA[(t + 1) & 1][0],
                            A + (size_t)rnext * KPAD, KPAD, n - rnext, KPAD, 16, KPAD);
                __builtin_amdgcn_s_wait_tensorcnt(1);   // tile t done, t+1 in flight
            } else {
                __builtin_amdgcn_s_wait_tensorcnt(0);
            }
        }
#else
        {   // fallback: cooperative 16B-chunk staging
            constexpr int CPR = KPAD / 8;
            bf16_t* dst = &sA[t & 1][0];
            for (int idx = tid; idx < 16 * CPR; idx += TPB) {
                int r = idx / CPR, c = idx - r * CPR;
                int gr = row0 + r;
                uint4 v = {0u, 0u, 0u, 0u};
                if (gr < n) v = *(const uint4*)(A + (size_t)gr * KPAD + c * 8);
                *(uint4*)(dst + (size_t)r * KPAD + c * 8) = v;
            }
        }
#endif
        __syncthreads();                            // tile t visible to all waves

        const bf16_t* tile = &sA[t & 1][0];
        v8f acc = {};
#pragma unroll
        for (int ks = 0; ks < KSTEPS; ++ks) {
            v16u a;
            const bf16_t* ap = tile + mlane * KPAD + ks * 32 + khalf * 8;
            a.h[0] = *(const v8bf*)(ap);            // k = ks*32 + khalf*8 .. +8
            a.h[1] = *(const v8bf*)(ap + 16);       // k = ks*32 + 16 + khalf*8 .. +8
            acc = __builtin_amdgcn_wmma_f32_16x16x32_bf16(
                /*neg_a=*/false, a.v, /*neg_b=*/false, bfrag[ks].v,
                /*c_mod=*/(short)0, acc, /*reuse_a=*/false, /*reuse_b=*/false);
        }

        const int base = row0 + (khalf << 3);
        if (row0 + 16 <= n) {                       // uniform: unguarded stores
#pragma unroll
            for (int r = 0; r < 8; ++r)
                C[(size_t)(base + r) * 128 + col] = acc[r];
        } else {
#pragma unroll
            for (int r = 0; r < 8; ++r)
                if (base + r < n) C[(size_t)(base + r) * 128 + col] = acc[r];
        }
        __syncthreads();                            // done reading sA[t&1] before reuse
    }
}

// ---------------- GCN aggregation (A_hat @ xw + b) --------------------------

// out[i,c] = 2*dinv[i]^2 * xw[i,c] + bias[c]   (also serves as scatter init)
__global__ void k_gcn_self_bias(const float* __restrict__ xw, const float* __restrict__ dinv,
                                const float* __restrict__ bias, float* __restrict__ out,
                                int n, int C) {
    long t = (long)blockIdx.x * blockDim.x + threadIdx.x;
    if (t >= (long)n * C) return;
    int i = (int)(t / C), c = (int)(t % C);
    float d = dinv[i];
    out[t] = 2.0f * d * d * xw[t] + bias[c];
}

// out[row] += dinv[row]*dinv[col] * xw[col]  over all edges, C=128 (float4/lane)
__global__ void k_gcn_edge_scatter128(const float* __restrict__ xw,
                                      const int* __restrict__ row, const int* __restrict__ col,
                                      const float* __restrict__ dinv,
                                      float* __restrict__ out, int E) {
    long t = (long)blockIdx.x * blockDim.x + threadIdx.x;
    if (t >= (long)E * 32) return;
    int e = (int)(t >> 5);
    int q = (int)((t & 31) << 2);
    int r = row[e], c = col[e];
    float coef = dinv[r] * dinv[c];
    const float4 v = *(const float4*)(xw + (size_t)c * 128 + q);
    float* o = out + (size_t)r * 128 + q;
    atomicAdd(o + 0, coef * v.x);
    atomicAdd(o + 1, coef * v.y);
    atomicAdd(o + 2, coef * v.z);
    atomicAdd(o + 3, coef * v.w);
}

// single-channel variant for the final projection
__global__ void k_gcn_edge_scatter1(const float* __restrict__ xw,
                                    const int* __restrict__ row, const int* __restrict__ col,
                                    const float* __restrict__ dinv,
                                    float* __restrict__ out, int E) {
    long t = (long)blockIdx.x * blockDim.x + threadIdx.x;
    if (t >= E) return;
    int r = row[t], c = col[t];
    atomicAdd(&out[r], dinv[r] * dinv[c] * xw[c]);
}

// ---------------- pooling / unpooling ---------------------------------------

__global__ void k_pool_scatter(const float* __restrict__ h, const int* __restrict__ cluster,
                               float* __restrict__ s, int n) {
    long t = (long)blockIdx.x * blockDim.x + threadIdx.x;
    if (t >= (long)n * 32) return;
    int i = (int)(t >> 5);
    int q = (int)((t & 31) << 2);
    int cl = cluster[i];
    const float4 v = *(const float4*)(h + (size_t)i * 128 + q);
    float* o = s + (size_t)cl * 128 + q;
    atomicAdd(o + 0, v.x);
    atomicAdd(o + 1, v.y);
    atomicAdd(o + 2, v.z);
    atomicAdd(o + 3, v.w);
}

__global__ void k_pool_div(float* __restrict__ s, const float* __restrict__ cnt, int nc) {
    long t = (long)blockIdx.x * blockDim.x + threadIdx.x;
    if (t >= (long)nc * 128) return;
    s[t] /= fmaxf(cnt[t >> 7], 1.0f);
}

__global__ void k_unpool128(const float* __restrict__ xc, const int* __restrict__ cluster,
                            float* __restrict__ xu, int n) {
    long t = (long)blockIdx.x * blockDim.x + threadIdx.x;
    if (t >= (long)n * 128) return;
    int i = (int)(t >> 7), c = (int)(t & 127);
    xu[t] = xc[(size_t)cluster[i] * 128 + c];
}

// ---------------- final (2*HID+IN) -> 1 projection (full f32) ---------------
// One wave per row, lanes partition the 264 columns, shuffle reduction.

__global__ void k_dot_final(const float* __restrict__ h0, const float* __restrict__ xin,
                            const float* __restrict__ xu, const float* __restrict__ W,
                            float* __restrict__ z, int n, int IN) {
    int r    = blockIdx.x * (TPB / 32) + (threadIdx.x >> 5);
    int lane = threadIdx.x & 31;
    if (r >= n) return;
    const float* a = h0 + (size_t)r * 128;
    const float* b = xin + (size_t)r * IN;
    const float* c = xu + (size_t)r * 128;
    float s = 0.0f;
#pragma unroll
    for (int k = lane; k < 128; k += 32) s += a[k] * W[k];
    if (lane < IN) s += b[lane] * W[128 + lane];
#pragma unroll
    for (int k = lane; k < 128; k += 32) s += c[k] * W[128 + IN + k];
#pragma unroll
    for (int off = 16; off > 0; off >>= 1) s += __shfl_down(s, off, 32);
    if (lane == 0) z[r] = s;
}

// ---------------- host orchestration ----------------------------------------

#define GRID(total) dim3((unsigned)(((long)(total) + TPB - 1) / TPB))

static void launch_gemm(hipStream_t stream, int Kpad, const bf16_t* A, const bf16_t* Wp,
                        float* C, int n) {
    dim3 g((n + 63) / 64);   // 4 row-tiles (64 rows) per block
    if (Kpad == 32)       k_wmma_gemm_bf16<32><<<g, TPB, 0, stream>>>(A, Wp, C, n);
    else if (Kpad == 128) k_wmma_gemm_bf16<128><<<g, TPB, 0, stream>>>(A, Wp, C, n);
    else                  k_wmma_gemm_bf16<256><<<g, TPB, 0, stream>>>(A, Wp, C, n);
}

// One GCN layer with 128 outputs: bf16 cast -> WMMA GEMM -> self+bias -> edge scatter
static void gcn128(hipStream_t stream,
                   const float* s0, int w0, const float* s1, int w1,
                   int Kpad, int n,
                   const float* Wf, int Krows, const float* bias,
                   const int* erow, const int* ecol, int E, const float* dinv,
                   bf16_t* Abf, bf16_t* Wbf, float* xw, float* out, int relu) {
    k_cast2_bf16<<<GRID((long)n * Kpad), TPB, 0, stream>>>(s0, w0, s1, w1, Abf, n, Kpad);
    k_packW_bf16<<<GRID((long)Kpad * 128), TPB, 0, stream>>>(Wf, Krows, Wbf, Kpad);
    launch_gemm(stream, Kpad, Abf, Wbf, xw, n);
    k_gcn_self_bias<<<GRID((long)n * 128), TPB, 0, stream>>>(xw, dinv, bias, out, n, 128);
    k_gcn_edge_scatter128<<<GRID((long)E * 32), TPB, 0, stream>>>(xw, erow, ecol, dinv, out, E);
    if (relu) k_relu_ip<<<GRID((long)n * 128), TPB, 0, stream>>>(out, (long)n * 128);
}

extern "C" void kernel_launch(void* const* d_in, const int* in_sizes, int n_in,
                              void* d_out, int out_size, void* d_ws, size_t ws_size,
                              hipStream_t stream) {
    const float* x   = (const float*)d_in[0];
    const int*   e0  = (const int*)d_in[1];
    const int*   cl0 = (const int*)d_in[2];
    const int*   cl1 = (const int*)d_in[3];
    const int*   cl2 = (const int*)d_in[4];
    const float* Wd0 = (const float*)d_in[5];   const float* bd0 = (const float*)d_in[6];
    const float* Wd1 = (const float*)d_in[7];   const float* bd1 = (const float*)d_in[8];
    const float* Wd2 = (const float*)d_in[9];   const float* bd2 = (const float*)d_in[10];
    const float* Wd3 = (const float*)d_in[11];  const float* bd3 = (const float*)d_in[12];
    const float* Wu0 = (const float*)d_in[13];  const float* bu0 = (const float*)d_in[14];
    const float* Wu1 = (const float*)d_in[15];  const float* bu1 = (const float*)d_in[16];
    const float* Wu2 = (const float*)d_in[17];  const float* bu2 = (const float*)d_in[18];

    const int E  = in_sizes[1] / 2;
    const int N0 = in_sizes[2];
    const int N1 = in_sizes[3];
    const int N2 = in_sizes[4];
    const int N3 = N0 / 8;
    const int IN = in_sizes[0] / N0;

    // ---- workspace carve (256B aligned) ----
    char*  ws  = (char*)d_ws;
    size_t off = 0;
    auto alloc = [&](size_t bytes) -> void* {
        void* p = ws + off;
        off = (off + bytes + 255) & ~(size_t)255;
        return p;
    };
    int*   e1    = (int*)alloc((size_t)2 * E * 4);
    int*   e2    = (int*)alloc((size_t)2 * E * 4);
    int*   e3    = (int*)alloc((size_t)2 * E * 4);
    float* dinv0 = (float*)alloc((size_t)N0 * 4);
    float* dinv1 = (float*)alloc((size_t)N1 * 4);
    float* dinv2 = (float*)alloc((size_t)N2 * 4);
    float* dinv3 = (float*)alloc((size_t)N3 * 4);
    float* cnt0  = (float*)alloc((size_t)N1 * 4);
    float* cnt1  = (float*)alloc((size_t)N2 * 4);
    float* cnt2  = (float*)alloc((size_t)N3 * 4);
    bf16_t* Abf  = (bf16_t*)alloc((size_t)N0 * 256 * 2);
    bf16_t* Wbf  = (bf16_t*)alloc((size_t)256 * 128 * 2);
    float* xw    = (float*)alloc((size_t)N0 * 128 * 4);
    float* h0    = (float*)alloc((size_t)N0 * 128 * 4);   // memory[0] (128 feat part)
    float* h1    = (float*)alloc((size_t)N1 * 128 * 4);   // memory[1]
    float* h2    = (float*)alloc((size_t)N2 * 128 * 4);   // memory[2]
    float* x1    = (float*)alloc((size_t)N1 * 128 * 4);
    float* x2    = (float*)alloc((size_t)N2 * 128 * 4);
    float* x3    = (float*)alloc((size_t)N3 * 128 * 4);
    float* h3    = (float*)alloc((size_t)N3 * 128 * 4);   // bottleneck GCN out
    float* xu    = (float*)alloc((size_t)N0 * 128 * 4);   // unpool buffer (reused)
    float* g     = (float*)alloc((size_t)N0 * 128 * 4);   // up-path GCN out (reused)
    float* z1    = (float*)alloc((size_t)N0 * 4);         // final x@W column
    float* z2    = (float*)alloc((size_t)N0 * 4);         // final pre-sigmoid
    (void)ws_size; (void)n_in; (void)out_size;

    // ---- static graph structure: coarse edges, dinv per level, cluster counts
    k_reindex<<<GRID(2L * E), TPB, 0, stream>>>(e0, cl0, e1, 2L * E);
    k_reindex<<<GRID(2L * E), TPB, 0, stream>>>(e1, cl1, e2, 2L * E);
    k_reindex<<<GRID(2L * E), TPB, 0, stream>>>(e2, cl2, e3, 2L * E);

    const int*  erows[4] = { e0, e1, e2, e3 };
    float*      dinvs[4] = { dinv0, dinv1, dinv2, dinv3 };
    const int   Ns[4]    = { N0, N1, N2, N3 };
    for (int l = 0; l < 4; ++l) {
        k_fill_f32<<<GRID(Ns[l]), TPB, 0, stream>>>(dinvs[l], 2.0f, Ns[l]);  // + 2I self-loops
        k_deg_count<<<GRID(E), TPB, 0, stream>>>(erows[l], dinvs[l], E);
        k_rsqrt_ip<<<GRID(Ns[l]), TPB, 0, stream>>>(dinvs[l], Ns[l]);
    }
    k_fill_f32<<<GRID(N1), TPB, 0, stream>>>(cnt0, 0.0f, N1);
    k_pool_count<<<GRID(N0), TPB, 0, stream>>>(cl0, cnt0, N0);
    k_fill_f32<<<GRID(N2), TPB, 0, stream>>>(cnt1, 0.0f, N2);
    k_pool_count<<<GRID(N1), TPB, 0, stream>>>(cl1, cnt1, N1);
    k_fill_f32<<<GRID(N3), TPB, 0, stream>>>(cnt2, 0.0f, N3);
    k_pool_count<<<GRID(N2), TPB, 0, stream>>>(cl2, cnt2, N2);

    // ---- down path ----
    gcn128(stream, x, IN, nullptr, 0, 32, N0, Wd0, IN, bd0,
           e0, e0 + E, E, dinv0, Abf, Wbf, xw, h0, 1);
    k_fill_f32<<<GRID((long)N1 * 128), TPB, 0, stream>>>(x1, 0.0f, (long)N1 * 128);
    k_pool_scatter<<<GRID((long)N0 * 32), TPB, 0, stream>>>(h0, cl0, x1, N0);
    k_pool_div<<<GRID((long)N1 * 128), TPB, 0, stream>>>(x1, cnt0, N1);

    gcn128(stream, x1, 128, nullptr, 0, 128, N1, Wd1, 128, bd1,
           e1, e1 + E, E, dinv1, Abf, Wbf, xw, h1, 1);
    k_fill_f32<<<GRID((long)N2 * 128), TPB, 0, stream>>>(x2, 0.0f, (long)N2 * 128);
    k_pool_scatter<<<GRID((long)N1 * 32), TPB, 0, stream>>>(h1, cl1, x2, N1);
    k_pool_div<<<GRID((long)N2 * 128), TPB, 0, stream>>>(x2, cnt1, N2);

    gcn128(stream, x2, 128, nullptr, 0, 128, N2, Wd2, 128, bd2,
           e2, e2 + E, E, dinv2, Abf, Wbf, xw, h2, 1);
    k_fill_f32<<<GRID((long)N3 * 128), TPB, 0, stream>>>(x3, 0.0f, (long)N3 * 128);
    k_pool_scatter<<<GRID((long)N2 * 32), TPB, 0, stream>>>(h2, cl2, x3, N2);
    k_pool_div<<<GRID((long)N3 * 128), TPB, 0, stream>>>(x3, cnt2, N3);

    // ---- bottleneck (no relu) ----
    gcn128(stream, x3, 128, nullptr, 0, 128, N3, Wd3, 128, bd3,
           e3, e3 + E, E, dinv3, Abf, Wbf, xw, h3, 0);

    // ---- up path ----
    // stage 0: level 2
    k_unpool128<<<GRID((long)N2 * 128), TPB, 0, stream>>>(h3, cl2, xu, N2);
    gcn128(stream, h2, 128, xu, 128, 256, N2, Wu0, 256, bu0,
           e2, e2 + E, E, dinv2, Abf, Wbf, xw, g, 1);
    // stage 1: level 1
    k_unpool128<<<GRID((long)N1 * 128), TPB, 0, stream>>>(g, cl1, xu, N1);
    gcn128(stream, h1, 128, xu, 128, 256, N1, Wu1, 256, bu1,
           e1, e1 + E, E, dinv1, Abf, Wbf, xw, g, 1);
    // stage 2: level 0, single output channel (full f32 path)
    k_unpool128<<<GRID((long)N0 * 128), TPB, 0, stream>>>(g, cl0, xu, N0);
    k_dot_final<<<GRID((long)N0 * 32), TPB, 0, stream>>>(h0, x, xu, Wu2, z1, N0, IN);
    k_gcn_self_bias<<<GRID(N0), TPB, 0, stream>>>(z1, dinv0, bu2, z2, N0, 1);
    k_gcn_edge_scatter1<<<GRID(E), TPB, 0, stream>>>(z1, e0, e0 + E, dinv0, z2, E);
    k_sigmoid<<<GRID(N0), TPB, 0, stream>>>(z2, (float*)d_out, N0);
}
